// SparsePagedAttention_27204322853266
// MI455X (gfx1250) — matrix operand order repaired and checked by reading, not compiled, not run on hardware
//
#include <hip/hip_runtime.h>
#include <hip/hip_bf16.h>

// ---------------------------------------------------------------------------
// Prompt-phase causal GQA flash-attention for gfx1250 (MI455X, wave32, WMMA)
//   B=2, S=2048, H=16, Hk=4, D=128, f32 I/O, f16 WMMA compute.
// Kernel 1: streaming flash attention, 64 q-rows per WG (4 waves x 16 rows),
//           64-key tiles staged in LDS, online softmax in exp2 domain.
// Kernel 2: score_sum over last 64 query rows using final m/l from workspace.
// ---------------------------------------------------------------------------

typedef __attribute__((ext_vector_type(16))) _Float16 v16h;
typedef __attribute__((ext_vector_type(8)))  float    v8f;

#define S_LEN    2048
#define NHEADS   16
#define NKV      4
#define HDIM     128
#define QSTRIDE  (NHEADS * HDIM)   // 2048 floats per token row of Q / Out
#define KVSTRIDE (NKV * HDIM)      // 512 floats per token row of K / V
#define NSCORE   64
#define NEG_INF  (-3.0e38f)

// SCALE * log2(e): do softmax in base-2 so exp2f -> v_exp_f32
__constant__ constexpr float kScaleLog2e = 0.08838834764831845f * 1.44269504088896340736f;

union Frag { v16h v; unsigned u[8]; };

__device__ __forceinline__ v8f wmma_f16(v16h a, v16h b, v8f c) {
  // 8 args: (neg_a, A, neg_b, B, c_mod, C, reuse_a, reuse_b)
  return __builtin_amdgcn_wmma_f32_16x16x32_f16(false, a, false, b, (short)0, c,
                                                false, false);
}

// A-fragment (16x32 f16) element K index for dword d, lane-half hf:
//   k = (d>>2)*16 + hf*8 + (d&3)*2   (ISA 7.12.2, 16-bit A 16x32)
// B-fragment (32x16 f16) element K index for dword d, lane-half hf:
//   k = (d>>2)*16 + hf*8 + (d&3)*2   (same order as A)

// ---------------------------------------------------------------------------
// Kernel 1: flash attention forward
// ---------------------------------------------------------------------------
__global__ __launch_bounds__(128) void attn_fwd_kernel(
    const float* __restrict__ Q, const float* __restrict__ K,
    const float* __restrict__ V, float* __restrict__ Out,
    float* __restrict__ wsM, float* __restrict__ wsL)
{
  // padded strides: all row strides are multiples of 4 bytes (b32 LDS access)
  __shared__ __align__(16) _Float16 sK[64][130];    // [key][dim]
  __shared__ __align__(16) _Float16 sVt[HDIM][66];  // [dim][key] (transposed)
  __shared__ __align__(16) _Float16 sP[4][16][68];  // per-wave P staging

  const int qt   = blockIdx.x;           // query tile (64 rows)
  const int h    = blockIdx.y;
  const int b    = blockIdx.z;
  const int hk   = h >> 2;               // GQA: 4 query heads per kv head
  const int q0   = qt * 64;
  const int tid  = threadIdx.x;
  const int wave = tid >> 5;
  const int lane = tid & 31;
  const int hf   = lane >> 4;            // lane half (wave32)
  const int ln   = lane & 15;

  // ---- load this wave's 16 Q rows as 4 A-fragments (f32 -> f16) ----
  Frag aQ[4];
  {
    const float* qr = Q + (size_t)(b * S_LEN + q0 + wave * 16 + ln) * QSTRIDE
                        + h * HDIM;
    #pragma unroll
    for (int c = 0; c < 4; ++c)
      #pragma unroll
      for (int d = 0; d < 8; ++d) {
        int kd = c * 32 + ((d >> 2) << 4) + (hf << 3) + ((d & 3) << 1);
        float2 f = *(const float2*)(qr + kd);
        aQ[c].v[2 * d]     = (_Float16)f.x;
        aQ[c].v[2 * d + 1] = (_Float16)f.y;
      }
  }

  v8f zero8 = {};
  v8f accO[8];
  #pragma unroll
  for (int dc = 0; dc < 8; ++dc) accO[dc] = zero8;
  float m_i[8], l_i[8];
  #pragma unroll
  for (int i = 0; i < 8; ++i) { m_i[i] = NEG_INF; l_i[i] = 0.f; }

  const int ntiles = qt + 1;  // causal: key tiles 0 .. qt
  const float* kbaseP = K + (size_t)(b * S_LEN) * KVSTRIDE + hk * HDIM + tid;
  const float* vbaseP = V + (size_t)(b * S_LEN) * KVSTRIDE + hk * HDIM + tid;

  for (int kt = 0; kt < ntiles; ++kt) {
    __syncthreads();  // previous iteration's LDS reads complete
    // ---- stage K (row-major) and V (transposed) tiles, f32 -> f16 ----
    {
      const float* kp = kbaseP + (size_t)(kt * 64) * KVSTRIDE;
      const float* vp = vbaseP + (size_t)(kt * 64) * KVSTRIDE;
      #pragma unroll 4
      for (int i = 0; i < 64; ++i) {
        sK[i][tid]  = (_Float16)kp[(size_t)i * KVSTRIDE];  // coalesced reads
        sVt[tid][i] = (_Float16)vp[(size_t)i * KVSTRIDE];
      }
      if (kt + 1 < ntiles) {  // global_prefetch_b8 the next tile
        __builtin_prefetch(kp + (size_t)64 * KVSTRIDE, 0, 1);
        __builtin_prefetch(vp + (size_t)64 * KVSTRIDE, 0, 1);
      }
    }
    __syncthreads();

    // ---- S = Q * K^T  (4 score tiles of 16x16, K-dim 128 = 4 wmma each) ----
    v8f st[4];
    #pragma unroll
    for (int t = 0; t < 4; ++t) {
      v8f acc = zero8;
      #pragma unroll
      for (int c = 0; c < 4; ++c) {
        Frag bK;
        #pragma unroll
        for (int d = 0; d < 8; ++d) {
          int kd = c * 32 + ((d >> 2) << 4) + (hf << 3) + ((d & 3) << 1);
          bK.u[d] = *(const unsigned*)&sK[t * 16 + ln][kd];
        }
        acc = wmma_f16(aQ[c].v, bK.v, acc);
      }
      #pragma unroll
      for (int i = 0; i < 8; ++i) st[t][i] = acc[i] * kScaleLog2e;
    }

    // ---- causal mask (only the diagonal tile can cross it) ----
    if (kt == ntiles - 1) {
      #pragma unroll
      for (int t = 0; t < 4; ++t)
        #pragma unroll
        for (int i = 0; i < 8; ++i) {
          int kg = kt * 64 + t * 16 + ln;            // C layout: col = ln
          int qg = q0 + wave * 16 + i + hf * 8;      //           row = i+8*hf
          if (kg > qg) st[t][i] = NEG_INF;
        }
    }

    // ---- online softmax (row stats per (i, hf); 16-lane shfl reductions) ---
    #pragma unroll
    for (int i = 0; i < 8; ++i) {
      float mx = fmaxf(fmaxf(st[0][i], st[1][i]), fmaxf(st[2][i], st[3][i]));
      #pragma unroll
      for (int mb = 1; mb < 16; mb <<= 1) mx = fmaxf(mx, __shfl_xor(mx, mb, 32));
      float mnew = fmaxf(m_i[i], mx);
      float rsc  = exp2f(m_i[i] - mnew);
      m_i[i] = mnew;
      float rs = 0.f;
      #pragma unroll
      for (int t = 0; t < 4; ++t) {
        float p = exp2f(st[t][i] - mnew);
        st[t][i] = p;
        rs += p;
      }
      #pragma unroll
      for (int mb = 1; mb < 16; mb <<= 1) rs += __shfl_xor(rs, mb, 32);
      l_i[i] = l_i[i] * rsc + rs;
      #pragma unroll
      for (int dc = 0; dc < 8; ++dc) accO[dc][i] *= rsc;
    }

    // ---- P: C layout -> LDS -> A layout (f16) ----
    #pragma unroll
    for (int t = 0; t < 4; ++t)
      #pragma unroll
      for (int i = 0; i < 8; ++i)
        sP[wave][i + hf * 8][t * 16 + ln] = (_Float16)st[t][i];
    // same-wave LDS RAW: DS ops are in-order, but fence the compiler + waitcnt
    asm volatile("s_wait_dscnt 0x0" ::: "memory");

    Frag aP[2];
    #pragma unroll
    for (int kk = 0; kk < 2; ++kk)
      #pragma unroll
      for (int d = 0; d < 8; ++d) {
        int kd = kk * 32 + ((d >> 2) << 4) + (hf << 3) + ((d & 3) << 1);
        aP[kk].u[d] = *(const unsigned*)&sP[wave][ln][kd];
      }

    // ---- O += P * V  (8 dim-chunks x 2 k-chunks) ----
    #pragma unroll
    for (int dc = 0; dc < 8; ++dc) {
      #pragma unroll
      for (int kk = 0; kk < 2; ++kk) {
        Frag bV;
        #pragma unroll
        for (int d = 0; d < 8; ++d) {
          int kd = kk * 32 + ((d >> 2) << 4) + (hf << 3) + ((d & 3) << 1);
          bV.u[d] = *(const unsigned*)&sVt[dc * 16 + ln][kd];
        }
        accO[dc] = wmma_f16(aP[kk].v, bV.v, accO[dc]);
      }
    }
  }

  // ---- epilogue: O /= l, store; last q-tile also spills m/l for kernel 2 ---
  #pragma unroll
  for (int i = 0; i < 8; ++i) {
    float inv  = 1.0f / l_i[i];
    size_t row = (size_t)(b * S_LEN + q0 + wave * 16 + i + hf * 8);
    float* op  = Out + row * QSTRIDE + h * HDIM + ln;
    #pragma unroll
    for (int dc = 0; dc < 8; ++dc) op[dc * 16] = accO[dc][i] * inv;
  }

  if (qt == (S_LEN / 64 - 1) && ln == 0) {
    int base = (b * NHEADS + h) * NSCORE;
    #pragma unroll
    for (int i = 0; i < 8; ++i) {
      int r = wave * 16 + i + hf * 8;
      wsM[base + r] = m_i[i];
      wsL[base + r] = l_i[i];
    }
  }
}

// ---------------------------------------------------------------------------
// Kernel 2: score_sum = sum over last 64 query rows of final softmax probs.
// One WG per (key-tile, h, b); disjoint output slices -> no global atomics.
// ---------------------------------------------------------------------------
__global__ __launch_bounds__(128) void score_sum_kernel(
    const float* __restrict__ Q, const float* __restrict__ K,
    const float* __restrict__ wsM, const float* __restrict__ wsL,
    float* __restrict__ Score)
{
  __shared__ __align__(16) _Float16 sK[64][130];
  __shared__ float sAcc[64];

  const int kt   = blockIdx.x;
  const int h    = blockIdx.y;
  const int b    = blockIdx.z;
  const int hk   = h >> 2;
  const int q0   = S_LEN - NSCORE;
  const int tid  = threadIdx.x;
  const int wave = tid >> 5;
  const int lane = tid & 31;
  const int hf   = lane >> 4;
  const int ln   = lane & 15;

  if (tid < 64) sAcc[tid] = 0.f;

  // Q fragments for the last 64 rows
  Frag aQ[4];
  {
    const float* qr = Q + (size_t)(b * S_LEN + q0 + wave * 16 + ln) * QSTRIDE
                        + h * HDIM;
    #pragma unroll
    for (int c = 0; c < 4; ++c)
      #pragma unroll
      for (int d = 0; d < 8; ++d) {
        int kd = c * 32 + ((d >> 2) << 4) + (hf << 3) + ((d & 3) << 1);
        float2 f = *(const float2*)(qr + kd);
        aQ[c].v[2 * d]     = (_Float16)f.x;
        aQ[c].v[2 * d + 1] = (_Float16)f.y;
      }
  }

  // final row stats from kernel 1
  float m_i[8], il_i[8];
  {
    int base = (b * NHEADS + h) * NSCORE;
    #pragma unroll
    for (int i = 0; i < 8; ++i) {
      int r = wave * 16 + i + hf * 8;
      m_i[i]  = wsM[base + r];
      il_i[i] = 1.0f / wsL[base + r];
    }
  }

  // stage the 64-key tile
  {
    const float* kp = K + (size_t)(b * S_LEN + kt * 64) * KVSTRIDE
                        + hk * HDIM + tid;
    #pragma unroll 4
    for (int i = 0; i < 64; ++i)
      sK[i][tid] = (_Float16)kp[(size_t)i * KVSTRIDE];
  }
  __syncthreads();

  v8f zero8 = {};
  #pragma unroll
  for (int t = 0; t < 4; ++t) {
    v8f acc = zero8;
    #pragma unroll
    for (int c = 0; c < 4; ++c) {
      Frag bK;
      #pragma unroll
      for (int d = 0; d < 8; ++d) {
        int kd = c * 32 + ((d >> 2) << 4) + (hf << 3) + ((d & 3) << 1);
        bK.u[d] = *(const unsigned*)&sK[t * 16 + ln][kd];
      }
      acc = wmma_f16(aQ[c].v, bK.v, acc);
    }
    float cs = 0.f;
    #pragma unroll
    for (int i = 0; i < 8; ++i) {
      float s2 = acc[i] * kScaleLog2e;
      int kg = kt * 64 + t * 16 + ln;
      int qg = q0 + wave * 16 + i + hf * 8;
      float p = (kg > qg) ? 0.f : exp2f(s2 - m_i[i]) * il_i[i];
      cs += p;  // partial column sum over this lane-half's 8 rows
    }
    cs += __shfl_xor(cs, 16, 32);  // combine lane halves (rows 0-7 + 8-15)
    if (hf == 0) atomicAdd(&sAcc[t * 16 + ln], cs);  // ds_add_f32 across waves
  }
  __syncthreads();

  if (tid < 64)
    Score[(size_t)(b * NHEADS + h) * S_LEN + kt * 64 + tid] = sAcc[tid];
}

// ---------------------------------------------------------------------------
extern "C" void kernel_launch(void* const* d_in, const int* in_sizes, int n_in,
                              void* d_out, int out_size, void* d_ws, size_t ws_size,
                              hipStream_t stream) {
  (void)in_sizes; (void)n_in; (void)out_size; (void)ws_size;
  const float* Q = (const float*)d_in[0];
  const float* K = (const float*)d_in[1];
  const float* V = (const float*)d_in[2];
  float* Out   = (float*)d_out;
  float* Score = Out + (size_t)2 * S_LEN * QSTRIDE;       // after [B*S, H*D]
  float* wsM   = (float*)d_ws;                            // [B*H*64] row max
  float* wsL   = wsM + 2 * NHEADS * NSCORE;               // [B*H*64] row sum

  dim3 grid(S_LEN / 64, NHEADS, 2);
  attn_fwd_kernel<<<grid, 128, 0, stream>>>(Q, K, V, Out, wsM, wsL);
  score_sum_kernel<<<grid, 128, 0, stream>>>(Q, K, wsM, wsL, Score);
}
